// WaveNet_89094801588557
// MI455X (gfx1250) — matrix operand order repaired and checked by reading, not compiled, not run on hardware
//
#include <hip/hip_runtime.h>

typedef __attribute__((ext_vector_type(16))) _Float16 v16h;
typedef __attribute__((ext_vector_type(8)))  float    v8f;

#define NBLK  5
#define DEPTH 10
#define NL    50          // layers
#define BATCH 4
#define LSEQ  8192
#define NT    (LSEQ/16)   // 512 column tiles per batch row
#define CS    512         // skip channels
#define CE    256         // post hidden
#define VOUT  256         // vocab

// --- fragment index helpers (wave32 WMMA f16; this file's contiguous k order, same for A and B) ---
// A matrix 16x32 f16: lane = h*16+r holds row r; element j -> k = h*16 + j
__device__ __host__ inline int a_frag_k(int h, int j) { return h * 16 + j; }
// B matrix 32x16 f16: lane (h,r) holds column r; element j -> k = h*16 + j
// C/D 16x16 f32: element v on lane (h,r) -> (m = v + 8h, n = r)

static __device__ inline v8f wmma_f16(v16h a, v16h b, v8f c) {
    return __builtin_amdgcn_wmma_f32_16x16x32_f16(false, a, false, b, (short)0, c, false, false);
}

// Hardware tanh (V_TANH_F32 is a CDNA5 TRANS op); sigmoid via tanh to avoid
// the branchy libm exp expansion in the hot layer kernel.
static __device__ inline float fast_tanh(float x) {
#if __has_builtin(__builtin_amdgcn_tanhf)
    return __builtin_amdgcn_tanhf(x);
#else
    return tanhf(x);
#endif
}
static __device__ inline float fast_sigmoid(float x) {
    return __builtin_fmaf(0.5f, fast_tanh(0.5f * x), 0.5f);
}

// ---------------------------------------------------------------------------
// Weight prep: scatter fp32 weights into fragment-ordered fp16 arrays.
// dst layout: [matrix][mtile][ktile][512], element e = lane*16 + j.
// ---------------------------------------------------------------------------
__global__ void prep_afrag(const float* __restrict__ src, _Float16* __restrict__ dst,
                           int mtiles, int ktiles, int rowstride, int count,
                           int mat_stride, int total) {
    int idx = blockIdx.x * blockDim.x + threadIdx.x;
    if (idx >= total) return;
    int per  = mtiles * ktiles * 512;
    int m_   = idx / per;
    int rest = idx - m_ * per;
    int tile = rest >> 9;
    int e    = rest & 511;
    int mt = tile / ktiles, kt = tile - mt * ktiles;
    int lane = e >> 4, j = e & 15;
    int h = lane >> 4, r = lane & 15;
    int row = mt * 16 + r;
    int k   = kt * 32 + a_frag_k(h, j);
    dst[idx] = (_Float16)src[(size_t)m_ * mat_stride + (size_t)row * rowstride + k];
}

// filt (rows 0..31) and gate (rows 32..63) stacked into one M=64, K=64 matrix per layer.
// K rows: k<32 -> tap0 (x[t-d], input chan k); k>=32 -> tap1 (x[t], input chan k-32).
__global__ void prep_fg(const float* __restrict__ filt, const float* __restrict__ gate,
                        _Float16* __restrict__ dst) {
    int idx = blockIdx.x * blockDim.x + threadIdx.x;
    const int total = NL * 8 * 512;           // 4 mtiles * 2 ktiles per layer
    if (idx >= total) return;
    int l    = idx / (8 * 512);
    int rest = idx - l * (8 * 512);
    int tile = rest >> 9;
    int e    = rest & 511;
    int mt = tile >> 1, kt = tile & 1;
    int lane = e >> 4, j = e & 15;
    int h = lane >> 4, r = lane & 15;
    int row = mt * 16 + r;                    // 0..63
    int k   = kt * 32 + a_frag_k(h, j);       // 0..63
    int tap = (k < 32) ? 0 : 1;
    int ci  = k & 31;
    float w;
    if (row < 32) w = filt[((size_t)(l * 32 + row) * 32 + ci) * 2 + tap];
    else          w = gate[((size_t)(l * 32 + (row - 32)) * 32 + ci) * 2 + tap];
    dst[idx] = (_Float16)w;
}

// ---------------------------------------------------------------------------
// Embedding + causal k=2 preprocess conv (Cr=32). VALU kernel, trivially small.
// ---------------------------------------------------------------------------
__global__ void embed_pre_kernel(const int* __restrict__ tokens, const float* __restrict__ emb,
                                 const float* __restrict__ prew, float* __restrict__ xout) {
    int idx = blockIdx.x * blockDim.x + threadIdx.x;
    if (idx >= BATCH * 32 * LSEQ) return;
    int t = idx & (LSEQ - 1);
    int c = (idx / LSEQ) & 31;
    int b = idx / (LSEQ * 32);
    int tok1 = tokens[b * LSEQ + t];
    const float* e1 = emb + tok1 * 32;
    const float* w  = prew + c * 64;          // [c][i][k], k fastest
    float acc = 0.f;
    if (t > 0) {
        int tok0 = tokens[b * LSEQ + t - 1];
        const float* e0 = emb + tok0 * 32;
        #pragma unroll
        for (int i = 0; i < 32; ++i) acc += w[i * 2 + 0] * e0[i] + w[i * 2 + 1] * e1[i];
    } else {
        #pragma unroll
        for (int i = 0; i < 32; ++i) acc += w[i * 2 + 1] * e1[i];
    }
    xout[idx] = acc;
}

// ---------------------------------------------------------------------------
// One WaveNet layer. Xcat staged to LDS via CDNA5 async global->LDS loads
// (ASYNCcnt path); causal-pad lanes zero-fill LDS under the divergent branch
// (EXEC masks the async op out for those lanes).
// ---------------------------------------------------------------------------
__global__ void __launch_bounds__(128)
layer_kernel(const float* __restrict__ xin, float* __restrict__ xout,
             const _Float16* __restrict__ wfg,   // [4 mt][2 kt][512]
             const _Float16* __restrict__ wres,  // [2 mt][512]
             _Float16* __restrict__ zout,        // [BATCH][NT][512] (B-frag order)
             int d) {
    __shared__ float    lds_x[4][16][64];        // [wave][t][k]  Xcat tile, fp32
    __shared__ _Float16 lds_z[4][16][32];        // [wave][t][c]

    int tid  = threadIdx.x;
    int wave = tid >> 5, lane = tid & 31;
    int h = lane >> 4, r = lane & 15;
    int gidx  = blockIdx.x;
    int b     = gidx / (NT / 4);
    int tile  = (gidx % (NT / 4)) * 4 + wave;
    int t0    = tile * 16;
    const float* xb = xin + (size_t)b * 32 * LSEQ;

    // cooperative stage of Xcat: rows 0..31 = x[:,t-d], rows 32..63 = x[:,t]
    for (int kk = h; kk < 64; kk += 2) {
        int c = kk & 31;
        int t = t0 + r - ((kk < 32) ? d : 0);
        if (t >= 0) {
            // async DMA: global -> LDS, no VGPR return (ASYNCcnt)
            unsigned laddr = (unsigned)(size_t)&lds_x[wave][r][kk];
            const float* gp = xb + (size_t)c * LSEQ + t;
            asm volatile("global_load_async_to_lds_b32 %0, %1, off"
                         :: "v"(laddr), "v"(gp) : "memory");
        } else {
            lds_x[wave][r][kk] = 0.0f;           // causal zero pad
        }
    }
    asm volatile("s_wait_asynccnt 0x0" ::: "memory");
    __syncthreads();

    // build B fragments (two K=32 steps) from LDS, fp32 -> fp16
    v16h xb0, xb1;
    const float* px = &lds_x[wave][r][0];
    #pragma unroll
    for (int j = 0; j < 16; ++j) {
        xb0[j] = (_Float16)px[h * 16 + j];
        xb1[j] = (_Float16)px[32 + h * 16 + j];
    }

    // filt+gate GEMM: M=64 (4 mtiles), K=64 (2 steps)
    v8f acc[4];
    #pragma unroll
    for (int mt = 0; mt < 4; ++mt) {
        #pragma unroll
        for (int v = 0; v < 8; ++v) acc[mt][v] = 0.0f;
        v16h a0 = *(const v16h*)(wfg + (mt * 2 + 0) * 512 + lane * 16);
        v16h a1 = *(const v16h*)(wfg + (mt * 2 + 1) * 512 + lane * 16);
        acc[mt] = wmma_f16(a0, xb0, acc[mt]);
        acc[mt] = wmma_f16(a1, xb1, acc[mt]);
    }

    // gated activation: z = tanh(f) * sigmoid(g); write to LDS in [t][c]
    #pragma unroll
    for (int mt = 0; mt < 2; ++mt)
        #pragma unroll
        for (int v = 0; v < 8; ++v) {
            float f = acc[mt][v], g = acc[mt + 2][v];
            float z = fast_tanh(f) * fast_sigmoid(g);
            lds_z[wave][r][mt * 16 + h * 8 + v] = (_Float16)z;
        }
    __syncthreads();

    // z as a B fragment (K=32); also store it verbatim for the batched skip GEMM
    v16h zb;
    #pragma unroll
    for (int j = 0; j < 16; ++j) zb[j] = lds_z[wave][r][h * 16 + j];
    *(v16h*)(zout + ((size_t)b * NT + tile) * 512 + lane * 16) = zb;

    // residual GEMM: x_new = res_w @ z + x_old (x_old preloaded into C)
    v8f racc[2];
    #pragma unroll
    for (int mt = 0; mt < 2; ++mt) {
        #pragma unroll
        for (int v = 0; v < 8; ++v)
            racc[mt][v] = lds_x[wave][r][32 + mt * 16 + h * 8 + v];
        v16h a = *(const v16h*)(wres + mt * 512 + lane * 16);
        racc[mt] = wmma_f16(a, zb, racc[mt]);
    }
    float* xo = xout + (size_t)b * 32 * LSEQ;
    #pragma unroll
    for (int mt = 0; mt < 2; ++mt)
        #pragma unroll
        for (int v = 0; v < 8; ++v) {
            int c = mt * 16 + h * 8 + v;
            xo[(size_t)c * LSEQ + t0 + r] = racc[mt][v];
        }
}

// ---------------------------------------------------------------------------
// Tail: skip = sum_l skip_w[l] @ z[l]  (one K=1600 GEMM, accumulators resident)
//       -> ReLU -> post_w1 (256x512) + b1 -> ReLU -> post_w2 (256x256) + b2.
// 8 waves / block, one 16-column tile per block.
// ---------------------------------------------------------------------------
__global__ void __launch_bounds__(256)
tail_kernel(const _Float16* __restrict__ zall,  // [NL][BATCH][NT][512]
            const _Float16* __restrict__ skw,   // [NL][32 mt][512]
            const _Float16* __restrict__ w1f,   // [16 mt][16 kt][512]
            const float* __restrict__ b1,
            const _Float16* __restrict__ w2f,   // [16 mt][8 kt][512]
            const float* __restrict__ b2,
            float* __restrict__ out, int Lout) {
    __shared__ _Float16 lds_s[16][CS];          // skip tile [t][c] fp16, post-ReLU
    __shared__ _Float16 lds_h[16][CE];          // hidden tile

    int tid  = threadIdx.x;
    int wave = tid >> 5, lane = tid & 31;
    int h = lane >> 4, r = lane & 15;
    int b    = blockIdx.x / NT;
    int tile = blockIdx.x % NT;
    int t0   = tile * 16;

    // ---- phase 1: skip projection, K batched over all 50 layers ----
    v8f sacc[4];
    #pragma unroll
    for (int q = 0; q < 4; ++q)
        #pragma unroll
        for (int v = 0; v < 8; ++v) sacc[q][v] = 0.0f;

    for (int l = 0; l < NL; ++l) {
        v16h zb = *(const v16h*)(zall + (((size_t)l * BATCH + b) * NT + tile) * 512 + lane * 16);
        const _Float16* wl = skw + (size_t)l * 32 * 512;
        if (l + 1 < NL) {
            // streaming weight prefetch -> global_prefetch_b8
            __builtin_prefetch(skw + (size_t)(l + 1) * 32 * 512 + (wave * 4) * 512 + lane * 16, 0, 1);
        }
        #pragma unroll
        for (int q = 0; q < 4; ++q) {
            int mt = wave * 4 + q;
            v16h a = *(const v16h*)(wl + mt * 512 + lane * 16);
            sacc[q] = wmma_f16(a, zb, sacc[q]);
        }
    }
    #pragma unroll
    for (int q = 0; q < 4; ++q) {
        int mt = wave * 4 + q;
        #pragma unroll
        for (int v = 0; v < 8; ++v) {
            float s = sacc[q][v];
            lds_s[r][mt * 16 + h * 8 + v] = (_Float16)(s > 0.f ? s : 0.f);
        }
    }
    __syncthreads();

    // ---- phase 2: h1 = relu(W1 @ skip + b1), W1 is 256x512 ----
    v8f hacc[2];
    #pragma unroll
    for (int q = 0; q < 2; ++q) {
        int mt = wave * 2 + q;
        #pragma unroll
        for (int v = 0; v < 8; ++v) hacc[q][v] = b1[mt * 16 + h * 8 + v];
    }
    #pragma unroll
    for (int ks = 0; ks < 16; ++ks) {
        v16h bb;
        #pragma unroll
        for (int j = 0; j < 16; ++j) bb[j] = lds_s[r][ks * 32 + h * 16 + j];
        #pragma unroll
        for (int q = 0; q < 2; ++q) {
            int mt = wave * 2 + q;
            v16h a = *(const v16h*)(w1f + ((size_t)(mt * 16 + ks)) * 512 + lane * 16);
            hacc[q] = wmma_f16(a, bb, hacc[q]);
        }
    }
    #pragma unroll
    for (int q = 0; q < 2; ++q) {
        int mt = wave * 2 + q;
        #pragma unroll
        for (int v = 0; v < 8; ++v) {
            float x = hacc[q][v];
            lds_h[r][mt * 16 + h * 8 + v] = (_Float16)(x > 0.f ? x : 0.f);
        }
    }
    __syncthreads();

    // ---- phase 3: out = W2 @ h1 + b2, W2 is 256x256 ----
    v8f oacc[2];
    #pragma unroll
    for (int q = 0; q < 2; ++q) {
        int mt = wave * 2 + q;
        #pragma unroll
        for (int v = 0; v < 8; ++v) oacc[q][v] = b2[mt * 16 + h * 8 + v];
    }
    #pragma unroll
    for (int ks = 0; ks < 8; ++ks) {
        v16h bb;
        #pragma unroll
        for (int j = 0; j < 16; ++j) bb[j] = lds_h[r][ks * 32 + h * 16 + j];
        #pragma unroll
        for (int q = 0; q < 2; ++q) {
            int mt = wave * 2 + q;
            v16h a = *(const v16h*)(w2f + ((size_t)(mt * 8 + ks)) * 512 + lane * 16);
            oacc[q] = wmma_f16(a, bb, oacc[q]);
        }
    }

    // trimmed store: reference keeps the LAST Lout columns
    int ot = t0 + r - (LSEQ - Lout);
    if (ot >= 0 && ot < Lout) {
        float* ob = out + (size_t)b * VOUT * Lout;
        #pragma unroll
        for (int q = 0; q < 2; ++q) {
            int mt = wave * 2 + q;
            #pragma unroll
            for (int v = 0; v < 8; ++v) {
                int row = mt * 16 + h * 8 + v;
                ob[(size_t)row * Lout + ot] = oacc[q][v];
            }
        }
    }
}

// ---------------------------------------------------------------------------
extern "C" void kernel_launch(void* const* d_in, const int* in_sizes, int n_in,
                              void* d_out, int out_size, void* d_ws, size_t ws_size,
                              hipStream_t stream) {
    (void)in_sizes; (void)n_in; (void)ws_size;
    const int*   tokens = (const int*)  d_in[0];
    const float* emb    = (const float*)d_in[1];
    const float* pre_w  = (const float*)d_in[2];
    const float* filt_w = (const float*)d_in[3];
    const float* gate_w = (const float*)d_in[4];
    const float* res_w  = (const float*)d_in[5];
    const float* skip_w = (const float*)d_in[6];
    const float* w1     = (const float*)d_in[7];
    const float* b1     = (const float*)d_in[8];
    const float* w2     = (const float*)d_in[9];
    const float* b2     = (const float*)d_in[10];
    float* out = (float*)d_out;
    int Lout = out_size / (BATCH * VOUT);

    char*  ws  = (char*)d_ws;
    size_t off = 0;
    auto alloc = [&](size_t bytes) {
        void* p = ws + off;
        off = (off + bytes + 255) & ~(size_t)255;
        return p;
    };
    float*    x0   = (float*)alloc((size_t)BATCH * 32 * LSEQ * 4);
    float*    x1   = (float*)alloc((size_t)BATCH * 32 * LSEQ * 4);
    _Float16* zall = (_Float16*)alloc((size_t)NL * BATCH * NT * 512 * 2);
    _Float16* wfg  = (_Float16*)alloc((size_t)NL * 8 * 512 * 2);
    _Float16* wres = (_Float16*)alloc((size_t)NL * 2 * 512 * 2);
    _Float16* wsk  = (_Float16*)alloc((size_t)NL * 32 * 512 * 2);
    _Float16* w1f  = (_Float16*)alloc((size_t)16 * 16 * 512 * 2);
    _Float16* w2f  = (_Float16*)alloc((size_t)16 * 8 * 512 * 2);

    // --- weight prep (fragment ordering) ---
    {
        int n = NL * 8 * 512;
        prep_fg<<<(n + 255) / 256, 256, 0, stream>>>(filt_w, gate_w, wfg);
    }
    { int n = NL * 2 * 512;
      prep_afrag<<<(n + 255) / 256, 256, 0, stream>>>(res_w,  wres, 2,  1, 32,  NL, 32 * 32,  n); }
    { int n = NL * 32 * 512;
      prep_afrag<<<(n + 255) / 256, 256, 0, stream>>>(skip_w, wsk, 32,  1, 32,  NL, 512 * 32, n); }
    { int n = 16 * 16 * 512;
      prep_afrag<<<(n + 255) / 256, 256, 0, stream>>>(w1,     w1f, 16, 16, 512, 1,  0,        n); }
    { int n = 16 * 8 * 512;
      prep_afrag<<<(n + 255) / 256, 256, 0, stream>>>(w2,     w2f, 16,  8, 256, 1,  0,        n); }

    // --- embedding + pre conv ---
    {
        int n = BATCH * 32 * LSEQ;
        embed_pre_kernel<<<(n + 255) / 256, 256, 0, stream>>>(tokens, emb, pre_w, x0);
    }

    // --- 50 dilated layers, ping-pong x, z written once in fragment order ---
    float* xin = x0;
    float* xo  = x1;
    int li = 0;
    for (int blk = 0; blk < NBLK; ++blk) {
        int d = 1;
        for (int dep = 0; dep < DEPTH; ++dep) {
            layer_kernel<<<BATCH * (NT / 4), 128, 0, stream>>>(
                xin, xo,
                wfg  + (size_t)li * 8 * 512,
                wres + (size_t)li * 2 * 512,
                zall + (size_t)li * BATCH * NT * 512,
                d);
            float* tmp = xin; xin = xo; xo = tmp;
            d *= 2; ++li;
        }
    }

    // --- fused skip-sum + post network ---
    tail_kernel<<<BATCH * NT, 256, 0, stream>>>(zall, wsk, w1f, b1, w2f, b2, out, Lout);
}
